// _MCLSTMCell_18494129177331
// MI455X (gfx1250) — matrix operand order repaired and checked by reading, not compiled, not run on hardware
//
#include <hip/hip_runtime.h>
#include <hip/hip_bf16.h>
#include <math.h>

// ---------------- problem constants ----------------
#define B_SZ   512
#define T_SZ   512
#define UNITS  64
#define AUXF   8
#define GFEAT  73          // 1 + 8 + 64 gate input features
#define KPAD   96          // K padded to 3 x 32 for bf16 WMMA
#define NCOLS  4224        // 64 (Wi) + 4096 (Wr) + 64 (Wo) concatenated output cols
#define NBLK   32          // persistent blocks, 16 batches each
#define BTILE  16
#define NTHREADS 256       // 8 wave32 waves
#define NWAVES 8

typedef __attribute__((ext_vector_type(16))) __bf16       v16bf;
typedef __attribute__((ext_vector_type(8)))  float        v8f;
typedef __attribute__((ext_vector_type(4)))  unsigned int u32x4;

union FragBF { v16bf v; u32x4 q[2]; };

// ---------------------------------------------------------------------------
// One gate "row group": 4 column tiles (64 cols), 3 WMMAs each, activation,
// row-wise L2 normalization (sum over the 64 cols via lane butterfly), then
// accumulate coef[row] * normalized value into msys accumulators.
// relu=true  -> r gate (coef = c[b,u])
// relu=false -> i gate (sigmoid, coef = xm[b])
// ---------------------------------------------------------------------------
__device__ __forceinline__ void gate_rows(
    const FragBF afr[3], const __bf16* __restrict__ Wb,
    const float* __restrict__ biasc, int tileBase,
    int l15, int lhalf, bool relu, const float coef[8], float msys[4][8])
{
    v8f accs[4];
    #pragma unroll
    for (int tv = 0; tv < 4; ++tv) {
        const int n = (tileBase + tv) * 16 + l15;     // output column for this lane
        const float bv = biasc[n];
        v8f acc;
        #pragma unroll
        for (int j = 0; j < 8; ++j) acc[j] = bv;      // bias broadcast (col-only)
        const __bf16* wrow = Wb + (size_t)n * KPAD + lhalf * 16;
        #pragma unroll
        for (int ch = 0; ch < 3; ++ch) {              // K = 96 in 3 chunks of 32
            FragBF bfr;
            bfr.q[0] = *(const u32x4*)(wrow + ch * 32);
            bfr.q[1] = *(const u32x4*)(wrow + ch * 32 + 8);
            acc = __builtin_amdgcn_wmma_f32_16x16x32_bf16(
                false, afr[ch].v, false, bfr.v, (short)0, acc, false, false);
        }
        accs[tv] = acc;
    }
    // activation + per-row sum of squares (row M = j + lhalf*8)
    float s[8];
    #pragma unroll
    for (int j = 0; j < 8; ++j) s[j] = 0.f;
    #pragma unroll
    for (int tv = 0; tv < 4; ++tv)
        #pragma unroll
        for (int j = 0; j < 8; ++j) {
            float v = accs[tv][j];
            v = relu ? fmaxf(v, 0.f) : 1.f / (1.f + __expf(-v));
            accs[tv][j] = v;
            s[j] += v * v;
        }
    // butterfly over the 16 lanes of each half = sum over the 64 cols of a row
    #pragma unroll
    for (int mask = 1; mask < 16; mask <<= 1)
        #pragma unroll
        for (int j = 0; j < 8; ++j) s[j] += __shfl_xor(s[j], mask, 32);
    #pragma unroll
    for (int j = 0; j < 8; ++j) {
        const float cf = coef[j] * rsqrtf(s[j]);      // h / ||h|| (ref has no eps)
        #pragma unroll
        for (int tv = 0; tv < 4; ++tv) msys[tv][j] += cf * accs[tv][j];
    }
}

// ---------------------------------------------------------------------------
// Init: fp32 weights -> bf16 concat matrix Wb[NCOLS][KPAD] (zero-padded K),
// concat bias, zero norm accumulators + grid-sync words.
// ---------------------------------------------------------------------------
__global__ __launch_bounds__(256) void mclstm_init(
    const float* __restrict__ Wi, const float* __restrict__ bi,
    const float* __restrict__ Wr, const float* __restrict__ br,
    const float* __restrict__ Wo, const float* __restrict__ bo,
    __bf16* __restrict__ Wb, float* __restrict__ biasc,
    float* __restrict__ norm_acc, unsigned* __restrict__ syncw)
{
    const size_t stride = (size_t)gridDim.x * blockDim.x;
    const size_t id0 = (size_t)blockIdx.x * blockDim.x + threadIdx.x;
    for (size_t e = id0; e < (size_t)NCOLS * KPAD; e += stride) {
        const int n = (int)(e / KPAD), k = (int)(e % KPAD);
        float v = 0.f;
        if (k < GFEAT) {
            if (n < UNITS)              v = Wi[k * UNITS + n];
            else if (n < UNITS + 4096)  v = Wr[(size_t)k * 4096 + (n - UNITS)];
            else                        v = Wo[k * UNITS + (n - (UNITS + 4096))];
        }
        Wb[e] = (__bf16)v;
    }
    for (size_t n = id0; n < NCOLS; n += stride)
        biasc[n] = (n < UNITS) ? bi[n]
                 : (n < UNITS + 4096) ? br[n - UNITS]
                 : bo[n - (UNITS + 4096)];
    if (id0 < T_SZ + 1) norm_acc[id0] = 0.f;
    if (id0 < 2)        syncw[id0] = 0u;
}

// ---------------------------------------------------------------------------
// Persistent scan kernel: 32 blocks x 16 batches; c lives in LDS for all 512
// steps; per step: build f (bf16), fused GEMM+act+norm+contraction via WMMA,
// elementwise update, global ||c||^2 accumulate, device-wide barrier.
// ---------------------------------------------------------------------------
__global__ __launch_bounds__(NTHREADS) void mclstm_scan(
    const float* __restrict__ xm, const float* __restrict__ xa,
    const __bf16* __restrict__ Wb, const float* __restrict__ biasc,
    float* __restrict__ norm_acc, unsigned* __restrict__ syncw,
    float* __restrict__ out_m, float* __restrict__ out_c)
{
    __shared__ __align__(16) __bf16 f_lds[BTILE][KPAD];
    __shared__ float c_lds[BTILE][UNITS];
    __shared__ float msys_lds[BTILE][UNITS];
    __shared__ float o_lds[BTILE][UNITS];
    __shared__ float xm_lds[BTILE];

    const int tid   = threadIdx.x;
    const int lane  = tid & 31;
    const int wv    = tid >> 5;
    const int l15   = lane & 15;
    const int lhalf = lane >> 4;
    const int b0    = blockIdx.x * BTILE;

    for (int i = tid; i < BTILE * UNITS; i += NTHREADS) ((float*)c_lds)[i] = 0.f;
    __syncthreads();

    for (int t = 0; t < T_SZ; ++t) {
        // ---- phase 1: scalar Frobenius norm + build f tile in bf16 ----
        const float ns = __hip_atomic_load(&norm_acc[t], __ATOMIC_RELAXED,
                                           __HIP_MEMORY_SCOPE_AGENT);
        const float invc = 1.f / (sqrtf(ns) + 1e-5f);
        for (int i = tid; i < BTILE * UNITS; i += NTHREADS) ((float*)msys_lds)[i] = 0.f;
        for (int i = tid; i < BTILE * KPAD; i += NTHREADS) {
            const int row = i / KPAD, col = i % KPAD;
            const int b = b0 + row;
            float v;
            if (col == 0)        { v = xm[(size_t)b * T_SZ + t]; xm_lds[row] = v; }
            else if (col <= AUXF)  v = xa[((size_t)b * T_SZ + t) * AUXF + (col - 1)];
            else if (col < GFEAT)  v = c_lds[row][col - 9] * invc;
            else                   v = 0.f;
            f_lds[row][col] = (__bf16)v;
        }
        __syncthreads();

        // ---- phase 2: A fragments from LDS (per-lane 2 x b128 per chunk) ----
        FragBF afr[3];
        #pragma unroll
        for (int ch = 0; ch < 3; ++ch) {
            const int kb = ch * 32 + lhalf * 16;
            afr[ch].q[0] = *(const u32x4*)&f_lds[l15][kb];
            afr[ch].q[1] = *(const u32x4*)&f_lds[l15][kb + 8];
        }

        float msys[4][8];
        #pragma unroll
        for (int tv = 0; tv < 4; ++tv)
            #pragma unroll
            for (int j = 0; j < 8; ++j) msys[tv][j] = 0.f;

        // ---- phase 3: r gate, 8 u-rows per wave (u = wv*8 .. wv*8+7) ----
        for (int ul = 0; ul < 8; ++ul) {
            const int u = wv * 8 + ul;
            float coef[8];
            #pragma unroll
            for (int j = 0; j < 8; ++j) coef[j] = c_lds[j + lhalf * 8][u];
            gate_rows(afr, Wb, biasc, 4 + 4 * u, l15, lhalf, true, coef, msys);
        }
        // ---- i gate on wave 0 (sigmoid + L2 norm, coef = xm) ----
        if (wv == 0) {
            float coef[8];
            #pragma unroll
            for (int j = 0; j < 8; ++j) coef[j] = xm_lds[j + lhalf * 8];
            gate_rows(afr, Wb, biasc, 0, l15, lhalf, false, coef, msys);
        }
        // ---- o gate on wave 1 (sigmoid, no norm) ----
        if (wv == 1) {
            #pragma unroll
            for (int tv = 0; tv < 4; ++tv) {
                const int n = (260 + tv) * 16 + l15;
                const float bv = biasc[n];
                v8f acc;
                #pragma unroll
                for (int j = 0; j < 8; ++j) acc[j] = bv;
                const __bf16* wrow = Wb + (size_t)n * KPAD + lhalf * 16;
                #pragma unroll
                for (int ch = 0; ch < 3; ++ch) {
                    FragBF bfr;
                    bfr.q[0] = *(const u32x4*)(wrow + ch * 32);
                    bfr.q[1] = *(const u32x4*)(wrow + ch * 32 + 8);
                    acc = __builtin_amdgcn_wmma_f32_16x16x32_bf16(
                        false, afr[ch].v, false, bfr.v, (short)0, acc, false, false);
                }
                #pragma unroll
                for (int j = 0; j < 8; ++j)
                    o_lds[j + lhalf * 8][tv * 16 + l15] = 1.f / (1.f + __expf(-acc[j]));
            }
        }
        // ---- flush per-wave partial m_in + m_sys into LDS ----
        #pragma unroll
        for (int tv = 0; tv < 4; ++tv)
            #pragma unroll
            for (int j = 0; j < 8; ++j)
                atomicAdd(&msys_lds[j + lhalf * 8][tv * 16 + l15], msys[tv][j]);
        __syncthreads();

        // ---- phase 5: elementwise update, outputs, ||c_new||^2 ----
        float sq = 0.f;
        #pragma unroll
        for (int e = 0; e < 4; ++e) {
            const int idx = tid * 4 + e;
            const int row = idx >> 6, v = idx & 63;
            const float m = msys_lds[row][v];
            const float o = o_lds[row][v];
            const float cn = (1.f - o) * m;
            const size_t oidx = ((size_t)t * B_SZ + b0 + row) * UNITS + v;
            out_m[oidx] = o * m;
            out_c[oidx] = cn;
            c_lds[row][v] = cn;
            sq += cn * cn;
        }
        #pragma unroll
        for (int mask = 1; mask < 32; mask <<= 1) sq += __shfl_xor(sq, mask, 32);
        if (lane == 0) atomicAdd(&norm_acc[t + 1], sq);

        // ---- device-wide barrier (global scalar norm couples all blocks) ----
        __threadfence();
        __syncthreads();
        if (tid == 0) {
            const unsigned target = (unsigned)(t + 1);
            const unsigned prev = __hip_atomic_fetch_add(
                &syncw[0], 1u, __ATOMIC_ACQ_REL, __HIP_MEMORY_SCOPE_AGENT);
            if (prev == (unsigned)NBLK * target - 1u) {
                __hip_atomic_store(&syncw[1], target, __ATOMIC_RELEASE,
                                   __HIP_MEMORY_SCOPE_AGENT);
            } else {
                while (__hip_atomic_load(&syncw[1], __ATOMIC_ACQUIRE,
                                         __HIP_MEMORY_SCOPE_AGENT) < target)
                    __builtin_amdgcn_s_sleep(2);
            }
        }
        __syncthreads();
    }
}

extern "C" void kernel_launch(void* const* d_in, const int* in_sizes, int n_in,
                              void* d_out, int out_size, void* d_ws, size_t ws_size,
                              hipStream_t stream)
{
    (void)in_sizes; (void)n_in; (void)out_size; (void)ws_size;
    const float* xm = (const float*)d_in[0];   // [512,512,1]
    const float* xa = (const float*)d_in[1];   // [512,512,8]
    const float* Wi = (const float*)d_in[2];   // [73,64]
    const float* bi = (const float*)d_in[3];   // [64]
    const float* Wr = (const float*)d_in[4];   // [73,4096]
    const float* br = (const float*)d_in[5];   // [4096]
    const float* Wo = (const float*)d_in[6];   // [73,64]
    const float* bo = (const float*)d_in[7];   // [64]

    float* out_m = (float*)d_out;                                  // [T,B,64]
    float* out_c = out_m + (size_t)T_SZ * B_SZ * UNITS;            // [T,B,64]

    char* ws = (char*)d_ws;
    __bf16* Wb = (__bf16*)ws;                                      // 811,008 B
    size_t off = (size_t)NCOLS * KPAD * sizeof(__bf16);
    float* biasc = (float*)(ws + off);   off += (size_t)NCOLS * sizeof(float);
    float* norm_acc = (float*)(ws + off); off += (size_t)(T_SZ + 1) * sizeof(float);
    off = (off + 63) & ~(size_t)63;
    unsigned* syncw = (unsigned*)(ws + off);

    mclstm_init<<<512, 256, 0, stream>>>(Wi, bi, Wr, br, Wo, bo,
                                         Wb, biasc, norm_acc, syncw);
    mclstm_scan<<<NBLK, NTHREADS, 0, stream>>>(xm, xa, Wb, biasc,
                                               norm_acc, syncw, out_m, out_c);
}
